// MolecularEncoder_50714973831823
// MI455X (gfx1250) — matrix operand order repaired and checked by reading, not compiled, not run on hardware
//
#include <hip/hip_runtime.h>
#include <cstdint>

typedef __bf16 bf16;
typedef __attribute__((ext_vector_type(16))) __bf16 v16bf;
typedef __attribute__((ext_vector_type(8)))  __bf16 v8bf;
typedef __attribute__((ext_vector_type(2)))  __bf16 v2bf;
typedef __attribute__((ext_vector_type(8)))  float  v8f;
typedef __attribute__((ext_vector_type(4)))  int    v4i;

#define DEVI __device__ __forceinline__

// gfx1250 async global->LDS path (ASYNCcnt), guarded so compile never breaks
#if __has_builtin(__builtin_amdgcn_global_load_async_to_lds_b128) && \
    __has_builtin(__builtin_amdgcn_s_wait_asynccnt)
#define HAVE_ASYNC_LDS 1
typedef __attribute__((address_space(1))) v4i gv4i;   // global AS int4
typedef __attribute__((address_space(3))) v4i sv4i;   // LDS AS int4
#else
#define HAVE_ASYNC_LDS 0
#endif

constexpr int cN = 25000, cE = 75000, cB = 512, cFA = 128, cFE = 32, cH = 512;
constexpr int cL = 5, cLOC = 512, cGLB = 256;

enum { F_SILU = 1, F_STF32 = 2, F_STBF16 = 4, F_SCATTER = 8, F_RES = 16 };

struct GemmArgs {
  const bf16* A0; const int* i0; int K0; int lda0;
  const bf16* A1; const int* i1; int K1; int lda1;
  const bf16* A2; const int* i2; int K2; int lda2;
  const bf16* Bw;
  const float* bias;
  const float* r1v; const float* r1n;   // rank-1 term: + r1v[row]*r1n[col]
  const float* resid;
  float* outF; bf16* outB;
  float* scat; const int* scatIdx;
  int M, Nt, flags;
};

DEVI float siluf(float x) { return x / (1.0f + __expf(-x)); }
DEVI float sigf(float x)  { return 1.0f / (1.0f + __expf(-x)); }

// ---------------------------------------------------------------------------
// Multi-stream WMMA GEMM: C[M,Nt] = act( sum_s gather(As,idx_s) @ Bw_rows + ...)
// 64x64 block tile, 128 threads (4 waves), each wave 32x32 via 4 WMMA tiles.
// A tile is staged with async global->LDS DMA (gfx1250) when available;
// B tile is transposed through registers (b128 loads, b32 K-pair stores).
// ---------------------------------------------------------------------------
__launch_bounds__(128)
__global__ void k_wmma_gemm(GemmArgs g) {
  __shared__ bf16 As[64][32];   // [m][k]
  __shared__ bf16 Bs[64][32];   // [n][k]  (K-contiguous for fragment packing)
  const int bm = blockIdx.x * 64;
  const int bn = blockIdx.y * 64;
  const int t = threadIdx.x;
  const int lane = t & 31, wave = t >> 5;
  const int wr = (wave >> 1) * 32, wc = (wave & 1) * 32;

  // staging coordinates
  const int rA = t >> 1, cA = (t & 1) << 4;       // A: row, 16-col half
  const int kB = (t >> 3) << 1;                   // B: k-pair row (0,2,..,30)
  const int nB = (t & 7) << 3;                    // B: 8-col group

  v8f acc[2][2] = {};

  const bf16* Aarr[3] = { g.A0, g.A1, g.A2 };
  const int*  Iarr[3] = { g.i0, g.i1, g.i2 };
  const int   Karr[3] = { g.K0, g.K1, g.K2 };
  const int   Larr[3] = { g.lda0, g.lda1, g.lda2 };

  int brow = 0;
  for (int s = 0; s < 3; ++s) {
    const bf16* A = Aarr[s];
    const int Ks = Karr[s];
    if (A == nullptr || Ks <= 0) continue;
    const int* idx = Iarr[s];
    const int lda = Larr[s];

    // hoist gather: one index load per stream, not per k-step
    const bf16* aptr = nullptr;
    {
      const int grow = bm + rA;
      if (grow < g.M) {
        const long arr = idx ? (long)idx[grow] : (long)grow;
        aptr = A + arr * (long)lda;
      }
    }
    const bf16* bbase = g.Bw + (long)brow * g.Nt + bn;

    for (int k0 = 0; k0 < Ks; k0 += 32) {
      __syncthreads();
      // ---- stage A tile ----
      if (aptr) {
#if HAVE_ASYNC_LDS
        // gfx1250 async DMA: global -> LDS without VGPR round-trip
        __builtin_amdgcn_global_load_async_to_lds_b128(
            (gv4i*)(aptr + k0 + cA),
            (sv4i*)&As[rA][cA], 0, 0);
        __builtin_amdgcn_global_load_async_to_lds_b128(
            (gv4i*)(aptr + k0 + cA + 8),
            (sv4i*)&As[rA][cA + 8], 0, 0);
#else
        v8bf a0 = *(const v8bf*)(aptr + k0 + cA);
        v8bf a1 = *(const v8bf*)(aptr + k0 + cA + 8);
        *(v8bf*)&As[rA][cA]     = a0;
        *(v8bf*)&As[rA][cA + 8] = a1;
#endif
        if (k0 + 32 < Ks) __builtin_prefetch(aptr + k0 + 32 + cA, 0, 3);
      } else {
        v8bf z = {};
        *(v8bf*)&As[rA][cA]     = z;
        *(v8bf*)&As[rA][cA + 8] = z;
      }
      // ---- stage B tile: load two k-rows, store K-pairs as b32 ----
      if (bn + nB < g.Nt) {
        const bf16* bp = bbase + (long)(k0 + kB) * g.Nt + nB;
        v8bf b0 = *(const v8bf*)(bp);
        v8bf b1 = *(const v8bf*)(bp + g.Nt);
        #pragma unroll
        for (int j2 = 0; j2 < 8; ++j2) {
          v2bf pr; pr[0] = b0[j2]; pr[1] = b1[j2];
          *(v2bf*)&Bs[nB + j2][kB] = pr;
        }
        if (k0 + 32 < Ks) __builtin_prefetch(bp + 32 * g.Nt, 0, 3);
      } else {
        v2bf z2 = {};
        #pragma unroll
        for (int j2 = 0; j2 < 8; ++j2) *(v2bf*)&Bs[nB + j2][kB] = z2;
      }
#if HAVE_ASYNC_LDS
      __builtin_amdgcn_s_wait_asynccnt(0);
#endif
      __syncthreads();

      const int hsel = lane >> 4, l16 = lane & 15;
      #pragma unroll
      for (int i = 0; i < 2; ++i) {
        v16bf a;
        const int ar = wr + i * 16 + l16;
        #pragma unroll
        for (int q = 0; q < 8; ++q) { // A frag: K pairs {0,1}..{6,7},{16,17}..{22,23}; +8 hi half
          const int kk = ((q < 4) ? (q << 1) : (16 + ((q - 4) << 1))) + (hsel << 3);
          a[2 * q] = As[ar][kk]; a[2 * q + 1] = As[ar][kk + 1];
        }
        #pragma unroll
        for (int j = 0; j < 2; ++j) {
          v16bf b;
          const int bc = wc + j * 16 + l16;
          #pragma unroll
          for (int q = 0; q < 8; ++q) { // B frag: K pairs {0,1}..{6,7},{16,17}..{22,23}; +8 hi half
            const int kk = ((q < 4) ? (q << 1) : (16 + ((q - 4) << 1))) + (hsel << 3);
            b[2 * q] = Bs[bc][kk]; b[2 * q + 1] = Bs[bc][kk + 1];
          }
          acc[i][j] = __builtin_amdgcn_wmma_f32_16x16x32_bf16(
              false, a, false, b, (short)0, acc[i][j], false, false);
        }
      }
    }
    brow += Ks;
  }

  // epilogue: C layout — reg r holds M = r + 8*(lane>=16), N = lane&15
  const int hsel = lane >> 4, l16 = lane & 15;
  #pragma unroll
  for (int i = 0; i < 2; ++i) {
    #pragma unroll
    for (int j = 0; j < 2; ++j) {
      const int col = bn + wc + j * 16 + l16;
      if (col >= g.Nt) continue;
      #pragma unroll
      for (int r = 0; r < 8; ++r) {
        const int row = bm + wr + i * 16 + (hsel << 3) + r;
        if (row >= g.M) continue;
        float v = acc[i][j][r];
        if (g.bias) v += g.bias[col];
        if (g.r1v)  v += g.r1v[row] * g.r1n[col];
        if (g.flags & F_SILU) v = siluf(v);
        if (g.flags & F_RES)  v += g.resid[(long)row * g.Nt + col];
        const long off = (long)row * g.Nt + col;
        if (g.flags & F_STF32)  g.outF[off] = v;
        if (g.flags & F_STBF16) g.outB[off] = (bf16)v;
        if (g.flags & F_SCATTER)
          unsafeAtomicAdd(&g.scat[(long)g.scatIdx[row] * g.Nt + col], v);
      }
    }
  }
}

// ---------------------------------------------------------------------------
// small helper kernels
// ---------------------------------------------------------------------------
__global__ void k_f2bf(const float* __restrict__ s, bf16* __restrict__ d, long n) {
  long i = (long)blockIdx.x * blockDim.x + threadIdx.x;
  if (i < n) d[i] = (bf16)s[i];
}
__global__ void k_vadd(const float* a, const float* b, float* o, int n) {
  int i = blockIdx.x * blockDim.x + threadIdx.x;
  if (i < n) o[i] = a[i] + b[i];
}
__global__ void k_deg(const int* __restrict__ dst, float* deg, int E) {
  int e = blockIdx.x * blockDim.x + threadIdx.x;
  if (e < E) unsafeAtomicAdd(&deg[dst[e]], 1.0f);
}
__global__ void k_edge_geom(const float* __restrict__ pos, const int* __restrict__ src,
                            const int* __restrict__ dst, float* dist, int E) {
  int e = blockIdx.x * blockDim.x + threadIdx.x;
  if (e >= E) return;
  int s = src[e], d = dst[e];
  float dx = pos[d * 3 + 0] - pos[s * 3 + 0];
  float dy = pos[d * 3 + 1] - pos[s * 3 + 1];
  float dz = pos[d * 3 + 2] - pos[s * 3 + 2];
  dist[e] = sqrtf(dx * dx + dy * dy + dz * dz);
}
__global__ void k_coef(const bf16* __restrict__ t, const float* __restrict__ w2,
                       const float* __restrict__ b2, float* coef, int n) {
  int row = blockIdx.x * (blockDim.x >> 5) + (threadIdx.x >> 5);
  int lane = threadIdx.x & 31;
  if (row >= n) return;
  float s = 0.f;
  for (int k = lane; k < 256; k += 32) s += (float)t[(long)row * 256 + k] * w2[k];
  for (int o = 16; o; o >>= 1) s += __shfl_xor(s, o, 32);
  if (lane == 0) coef[row] = s + b2[0];
}
__global__ void k_pos_edge(const float* __restrict__ pos, const int* __restrict__ src,
                           const int* __restrict__ dst, const float* __restrict__ dist,
                           const float* __restrict__ coef, float* acc, int E) {
  int e = blockIdx.x * blockDim.x + threadIdx.x;
  if (e >= E) return;
  int s = src[e], d = dst[e];
  float dn = fmaxf(dist[e], 1e-6f);
  float c = coef[d];
  #pragma unroll
  for (int k = 0; k < 3; ++k) {
    float dir = (pos[s * 3 + k] - pos[d * 3 + k]) / dn;
    unsafeAtomicAdd(&acc[d * 3 + k], c * dir);
  }
}
__global__ void k_pos_fin(float* pos, const float* acc, const float* deg, int n3) {
  int i = blockIdx.x * blockDim.x + threadIdx.x;
  if (i < n3) pos[i] += acc[i] / fmaxf(deg[i / 3], 1.0f);
}
__global__ void k_attn_score(const float* __restrict__ kf, const float* __restrict__ q,
                             const int* __restrict__ batch, float* ae, float* ssum, int n) {
  int i = blockIdx.x * blockDim.x + threadIdx.x;
  if (i >= n * 8) return;
  int a = i >> 3, hh = i & 7;
  float s = 0.f;
  #pragma unroll 4
  for (int d = 0; d < 64; ++d) s += kf[(long)a * 512 + hh * 64 + d] * q[hh * 64 + d];
  float e = __expf(s * 0.125f);
  ae[i] = e;
  unsafeAtomicAdd(&ssum[batch[a] * 8 + hh], e);
}
__global__ void k_attn_pool(const float* __restrict__ vf, const float* __restrict__ ae,
                            const float* __restrict__ ssum, const int* __restrict__ batch,
                            float* pooled, int n) {
  long i = (long)blockIdx.x * blockDim.x + threadIdx.x;
  if (i >= (long)n * 512) return;
  int a = (int)(i >> 9), d = (int)(i & 511), hh = d >> 6;
  int b = batch[a];
  float an = ae[a * 8 + hh] / (ssum[b * 8 + hh] + 1e-8f);
  unsafeAtomicAdd(&pooled[(long)b * 512 + d], an * vf[i]);
}
__global__ void k_lstm(const float* __restrict__ gates, float* hc, float* cc, bf16* hcb, int n) {
  int i = blockIdx.x * blockDim.x + threadIdx.x;
  if (i >= n) return;
  int b = i >> 9, j = i & 511;
  const float* g = gates + (long)b * 2048;
  float ig = sigf(g[j]);
  float fg = sigf(g[512 + j]);
  float gg = tanhf(g[1024 + j]);
  float og = sigf(g[1536 + j]);
  float c = fg * cc[i] + ig * gg;
  float h = og * tanhf(c);
  cc[i] = c; hc[i] = h; hcb[i] = (bf16)h;
}
__global__ void k_s2s_score(const float* __restrict__ h, const float* __restrict__ hc,
                            const int* __restrict__ batch, float* ae, float* ssum, int n) {
  int row = blockIdx.x * (blockDim.x >> 5) + (threadIdx.x >> 5);
  int lane = threadIdx.x & 31;
  if (row >= n) return;
  int b = batch[row];
  float s = 0.f;
  for (int j = lane; j < 512; j += 32) s += h[(long)row * 512 + j] * hc[(long)b * 512 + j];
  for (int o = 16; o; o >>= 1) s += __shfl_xor(s, o, 32);
  if (lane == 0) {
    float e = __expf(s);
    ae[row] = e;
    unsafeAtomicAdd(&ssum[b], e);
  }
}
__global__ void k_s2s_pool(const float* __restrict__ h, const float* __restrict__ ae,
                           const float* __restrict__ ssum, const int* __restrict__ batch,
                           float* r, int n) {
  long i = (long)blockIdx.x * blockDim.x + threadIdx.x;
  if (i >= (long)n * 512) return;
  int a = (int)(i >> 9), j = (int)(i & 511);
  int b = batch[a];
  float an = ae[a] / (ssum[b] + 1e-8f);
  unsafeAtomicAdd(&r[(long)b * 512 + j], an * h[i]);
}
__global__ void k_qstar(const float* __restrict__ hc, const float* __restrict__ r,
                        float* q, bf16* qb, int n) {
  int i = blockIdx.x * blockDim.x + threadIdx.x;
  if (i >= n) return;
  int b = i >> 10, j = i & 1023;
  float v = (j < 512) ? hc[b * 512 + j] : r[b * 512 + (j - 512)];
  q[i] = v; qb[i] = (bf16)v;
}

// ---------------------------------------------------------------------------
extern "C" void kernel_launch(void* const* d_in, const int* in_sizes, int n_in,
                              void* d_out, int out_size, void* d_ws, size_t ws_size,
                              hipStream_t stream) {
  (void)in_sizes; (void)n_in; (void)out_size; (void)ws_size;
  // inputs
  const float* x      = (const float*)d_in[0];
  const int*   eidx   = (const int*)d_in[1];
  const int*   srcI   = eidx;          // edge_index[0]
  const int*   dstI   = eidx + cE;     // edge_index[1]
  const int*   batch  = (const int*)d_in[2];
  const float* eattr  = (const float*)d_in[3];
  const float* posIn  = (const float*)d_in[4];
  const float* W_atom = (const float*)d_in[5];  const float* b_atom = (const float*)d_in[6];
  const float* W_edge = (const float*)d_in[7];  const float* b_edge = (const float*)d_in[8];
  const float* msg_w1 = (const float*)d_in[9];  const float* msg_b1 = (const float*)d_in[10];
  const float* msg_w2 = (const float*)d_in[11]; const float* msg_b2 = (const float*)d_in[12];
  const float* upd_w1 = (const float*)d_in[13]; const float* upd_b1 = (const float*)d_in[14];
  const float* upd_w2 = (const float*)d_in[15]; const float* upd_b2 = (const float*)d_in[16];
  const float* pos_w1 = (const float*)d_in[17]; const float* pos_b1 = (const float*)d_in[18];
  const float* pos_w2 = (const float*)d_in[19]; const float* pos_b2 = (const float*)d_in[20];
  const float* attn_q = (const float*)d_in[21];
  const float* W_key  = (const float*)d_in[22]; const float* b_key  = (const float*)d_in[23];
  const float* W_val  = (const float*)d_in[24]; const float* b_val  = (const float*)d_in[25];
  const float* l_wih  = (const float*)d_in[26]; const float* l_bih  = (const float*)d_in[27];
  const float* l_whh  = (const float*)d_in[28]; const float* l_bhh  = (const float*)d_in[29];
  const float* lp_w1  = (const float*)d_in[30]; const float* lp_b1  = (const float*)d_in[31];
  const float* lp_w2  = (const float*)d_in[32]; const float* lp_b2  = (const float*)d_in[33];
  const float* gp_w1  = (const float*)d_in[34]; const float* gp_b1  = (const float*)d_in[35];
  const float* gp_w2  = (const float*)d_in[36]; const float* gp_b2  = (const float*)d_in[37];

  float* zloc  = (float*)d_out;
  float* zglob = zloc + (long)cB * cLOC;

  // bump allocator over d_ws
  uintptr_t cur = (uintptr_t)d_ws;
  auto alloc = [&](size_t bytes) -> void* {
    cur = (cur + 255) & ~(uintptr_t)255;
    void* p = (void*)cur; cur += bytes; return p;
  };

  // bf16 weight arena
  bf16* Watom_b = (bf16*)alloc((long)cFA * cH * 2);
  bf16* Wedge_b = (bf16*)alloc((long)cFE * cFE * 2);
  bf16 *msgW_b[cL], *msg2W_b[cL], *updW1_b[cL], *updW2_b[cL], *posW1_b[cL];
  for (int l = 0; l < cL; ++l) {
    msgW_b[l]  = (bf16*)alloc((long)(1024 + 32) * 512 * 2);
    msg2W_b[l] = (bf16*)alloc((long)512 * 512 * 2);
    updW1_b[l] = (bf16*)alloc((long)1024 * 512 * 2);
    updW2_b[l] = (bf16*)alloc((long)512 * 512 * 2);
    posW1_b[l] = (bf16*)alloc((long)512 * 256 * 2);
  }
  bf16* Wkey_b  = (bf16*)alloc((long)512 * 512 * 2);
  bf16* Wval_b  = (bf16*)alloc((long)512 * 512 * 2);
  bf16* lstmW_b = (bf16*)alloc((long)1536 * 2048 * 2);
  bf16* lpW1_b  = (bf16*)alloc((long)512 * 512 * 2);
  bf16* lpW2_b  = (bf16*)alloc((long)512 * 512 * 2);
  bf16* gpW1_b  = (bf16*)alloc((long)1024 * 512 * 2);
  bf16* gpW2_b  = (bf16*)alloc((long)512 * 256 * 2);

  // activations
  bf16*  x_b     = (bf16*)alloc((long)cN * cFA * 2);
  bf16*  eattr_b = (bf16*)alloc((long)cE * cFE * 2);
  bf16*  ea_b    = (bf16*)alloc((long)cE * cFE * 2);
  float* h       = (float*)alloc((long)cN * cH * 4);
  bf16*  h_b     = (bf16*)alloc((long)cN * cH * 2);
  float* agg     = (float*)alloc((long)cN * cH * 4);
  bf16*  agg_b   = (bf16*)alloc((long)cN * cH * 2);
  bf16*  m1_b    = (bf16*)alloc((long)cE * cH * 2);
  bf16*  u_b     = (bf16*)alloc((long)cN * cH * 2);
  bf16*  t_b     = (bf16*)alloc((long)cN * 256 * 2);
  float* coef    = (float*)alloc((long)cN * 4);
  float* dist    = (float*)alloc((long)cE * 4);
  float* deg     = (float*)alloc((long)cN * 4);
  float* pos_c   = (float*)alloc((long)cN * 3 * 4);
  float* pos_a   = (float*)alloc((long)cN * 3 * 4);
  float* kf      = (float*)alloc((long)cN * cH * 4);
  float* vf      = (float*)alloc((long)cN * cH * 4);
  float* ae_at   = (float*)alloc((long)cN * 8 * 4);
  float* s_at    = (float*)alloc((long)cB * 8 * 4);
  float* pooled  = (float*)alloc((long)cB * cH * 4);
  bf16*  pooled_b= (bf16*)alloc((long)cB * cH * 2);
  float* hc      = (float*)alloc((long)cB * cH * 4);
  float* ccb     = (float*)alloc((long)cB * cH * 4);
  bf16*  hc_b    = (bf16*)alloc((long)cB * cH * 2);
  float* rpool   = (float*)alloc((long)cB * cH * 4);
  float* q_star  = (float*)alloc((long)cB * 1024 * 4);
  bf16*  qstar_b = (bf16*)alloc((long)cB * 1024 * 2);
  float* gates   = (float*)alloc((long)cB * 2048 * 4);
  float* ae_s    = (float*)alloc((long)cN * 4);
  float* s_s     = (float*)alloc((long)cB * 4);
  float* b_lstm  = (float*)alloc(2048 * 4);
  bf16*  t1_b    = (bf16*)alloc((long)cB * 512 * 2);
  bf16*  t2_b    = (bf16*)alloc((long)cB * 512 * 2);

  auto f2bf = [&](const float* s, bf16* d, long n) {
    k_f2bf<<<dim3((unsigned)((n + 255) / 256)), 256, 0, stream>>>(s, d, n);
  };
  auto gemm = [&](const bf16* A0, const int* i0, int K0, int lda0,
                  const bf16* A1, const int* i1, int K1, int lda1,
                  const bf16* A2, const int* i2, int K2, int lda2,
                  const bf16* Bw, const float* bias,
                  const float* r1v, const float* r1n, const float* resid,
                  float* outF, bf16* outB, float* scat, const int* scatIdx,
                  int M, int Nt, int flags) {
    GemmArgs g{};
    g.A0 = A0; g.i0 = i0; g.K0 = K0; g.lda0 = lda0;
    g.A1 = A1; g.i1 = i1; g.K1 = K1; g.lda1 = lda1;
    g.A2 = A2; g.i2 = i2; g.K2 = K2; g.lda2 = lda2;
    g.Bw = Bw; g.bias = bias; g.r1v = r1v; g.r1n = r1n; g.resid = resid;
    g.outF = outF; g.outB = outB; g.scat = scat; g.scatIdx = scatIdx;
    g.M = M; g.Nt = Nt; g.flags = flags;
    dim3 grid((M + 63) / 64, (Nt + 63) / 64);
    k_wmma_gemm<<<grid, 128, 0, stream>>>(g);
  };

  // ---- weight conversion (deterministic, every call) ----
  f2bf(W_atom, Watom_b, (long)cFA * cH);
  f2bf(W_edge, Wedge_b, (long)cFE * cFE);
  for (int l = 0; l < cL; ++l) {
    f2bf(msg_w1 + (long)l * 1057 * 512,          msgW_b[l],               (long)1024 * 512);
    f2bf(msg_w1 + ((long)l * 1057 + 1025) * 512, msgW_b[l] + 1024 * 512,  (long)32 * 512);
    f2bf(msg_w2 + (long)l * 512 * 512,  msg2W_b[l], (long)512 * 512);
    f2bf(upd_w1 + (long)l * 1024 * 512, updW1_b[l], (long)1024 * 512);
    f2bf(upd_w2 + (long)l * 512 * 512,  updW2_b[l], (long)512 * 512);
    f2bf(pos_w1 + (long)l * 512 * 256,  posW1_b[l], (long)512 * 256);
  }
  f2bf(W_key, Wkey_b, (long)512 * 512);
  f2bf(W_val, Wval_b, (long)512 * 512);
  f2bf(l_wih, lstmW_b, (long)1024 * 2048);
  f2bf(l_whh, lstmW_b + (long)1024 * 2048, (long)512 * 2048);
  f2bf(lp_w1, lpW1_b, (long)512 * 512);
  f2bf(lp_w2, lpW2_b, (long)512 * 512);
  f2bf(gp_w1, gpW1_b, (long)1024 * 512);
  f2bf(gp_w2, gpW2_b, (long)512 * 256);
  f2bf(x, x_b, (long)cN * cFA);
  f2bf(eattr, eattr_b, (long)cE * cFE);
  k_vadd<<<8, 256, 0, stream>>>(l_bih, l_bhh, b_lstm, 2048);

  // ---- graph setup ----
  (void)hipMemcpyAsync(pos_c, posIn, (long)cN * 3 * 4, hipMemcpyDeviceToDevice, stream);
  (void)hipMemsetAsync(deg, 0, (long)cN * 4, stream);
  k_deg<<<(cE + 255) / 256, 256, 0, stream>>>(dstI, deg, cE);

  // h = x @ W_atom + b_atom ; ea = edge_attr @ W_edge + b_edge
  gemm(x_b, nullptr, cFA, cFA, nullptr, nullptr, 0, 0, nullptr, nullptr, 0, 0,
       Watom_b, b_atom, nullptr, nullptr, nullptr,
       h, h_b, nullptr, nullptr, cN, cH, F_STF32 | F_STBF16);
  gemm(eattr_b, nullptr, cFE, cFE, nullptr, nullptr, 0, 0, nullptr, nullptr, 0, 0,
       Wedge_b, b_edge, nullptr, nullptr, nullptr,
       nullptr, ea_b, nullptr, nullptr, cE, cFE, F_STBF16);

  // ---- L message-passing layers ----
  for (int l = 0; l < cL; ++l) {
    k_edge_geom<<<(cE + 255) / 256, 256, 0, stream>>>(pos_c, srcI, dstI, dist, cE);
    (void)hipMemsetAsync(agg, 0, (long)cN * cH * 4, stream);

    // m1 = silu( [h[dst], h[src], dist, ea] @ msg_w1 + msg_b1 )  (dist as rank-1)
    gemm(h_b, dstI, cH, cH, h_b, srcI, cH, cH, ea_b, nullptr, cFE, cFE,
         msgW_b[l], msg_b1 + (long)l * cH,
         dist, msg_w1 + ((long)l * 1057 + 1024) * 512, nullptr,
         nullptr, m1_b, nullptr, nullptr, cE, cH, F_SILU | F_STBF16);
    // agg[dst] += silu( m1 @ msg_w2 + msg_b2 )   (fused segment_sum)
    gemm(m1_b, nullptr, cH, cH, nullptr, nullptr, 0, 0, nullptr, nullptr, 0, 0,
         msg2W_b[l], msg_b2 + (long)l * cH, nullptr, nullptr, nullptr,
         nullptr, nullptr, agg, dstI, cE, cH, F_SILU | F_SCATTER);
    f2bf(agg, agg_b, (long)cN * cH);

    // u = silu([h, agg] @ upd_w1 + b1)
    gemm(h_b, nullptr, cH, cH, agg_b, nullptr, cH, cH, nullptr, nullptr, 0, 0,
         updW1_b[l], upd_b1 + (long)l * cH, nullptr, nullptr, nullptr,
         nullptr, u_b, nullptr, nullptr, cN, cH, F_SILU | F_STBF16);
    // pos coef path (uses agg of this layer)
    gemm(agg_b, nullptr, cH, cH, nullptr, nullptr, 0, 0, nullptr, nullptr, 0, 0,
         posW1_b[l], pos_b1 + (long)l * 256, nullptr, nullptr, nullptr,
         nullptr, t_b, nullptr, nullptr, cN, 256, F_SILU | F_STBF16);
    // h = h + (u @ upd_w2 + b2)
    gemm(u_b, nullptr, cH, cH, nullptr, nullptr, 0, 0, nullptr, nullptr, 0, 0,
         updW2_b[l], upd_b2 + (long)l * cH, nullptr, nullptr, h,
         h, h_b, nullptr, nullptr, cN, cH, F_RES | F_STF32 | F_STBF16);

    k_coef<<<(cN + 3) / 4, 128, 0, stream>>>(t_b, pos_w2 + (long)l * 256,
                                             pos_b2 + l, coef, cN);
    (void)hipMemsetAsync(pos_a, 0, (long)cN * 3 * 4, stream);
    k_pos_edge<<<(cE + 255) / 256, 256, 0, stream>>>(pos_c, srcI, dstI, dist, coef, pos_a, cE);
    k_pos_fin<<<(cN * 3 + 255) / 256, 256, 0, stream>>>(pos_c, pos_a, deg, cN * 3);
  }

  // ---- attention pooling ----
  gemm(h_b, nullptr, cH, cH, nullptr, nullptr, 0, 0, nullptr, nullptr, 0, 0,
       Wkey_b, b_key, nullptr, nullptr, nullptr,
       kf, nullptr, nullptr, nullptr, cN, cH, F_STF32);
  gemm(h_b, nullptr, cH, cH, nullptr, nullptr, 0, 0, nullptr, nullptr, 0, 0,
       Wval_b, b_val, nullptr, nullptr, nullptr,
       vf, nullptr, nullptr, nullptr, cN, cH, F_STF32);
  (void)hipMemsetAsync(s_at, 0, (long)cB * 8 * 4, stream);
  (void)hipMemsetAsync(pooled, 0, (long)cB * cH * 4, stream);
  k_attn_score<<<(cN * 8 + 255) / 256, 256, 0, stream>>>(kf, attn_q, batch, ae_at, s_at, cN);
  k_attn_pool<<<(unsigned)(((long)cN * 512 + 255) / 256), 256, 0, stream>>>(
      vf, ae_at, s_at, batch, pooled, cN);
  f2bf(pooled, pooled_b, (long)cB * cH);

  // ---- Set2Set ----
  (void)hipMemsetAsync(hc, 0, (long)cB * cH * 4, stream);
  (void)hipMemsetAsync(ccb, 0, (long)cB * cH * 4, stream);
  (void)hipMemsetAsync(hc_b, 0, (long)cB * cH * 2, stream);
  (void)hipMemsetAsync(q_star, 0, (long)cB * 1024 * 4, stream);
  (void)hipMemsetAsync(qstar_b, 0, (long)cB * 1024 * 2, stream);
  for (int step = 0; step < 3; ++step) {
    gemm(qstar_b, nullptr, 1024, 1024, hc_b, nullptr, 512, 512, nullptr, nullptr, 0, 0,
         lstmW_b, b_lstm, nullptr, nullptr, nullptr,
         gates, nullptr, nullptr, nullptr, cB, 2048, F_STF32);
    k_lstm<<<(cB * 512 + 255) / 256, 256, 0, stream>>>(gates, hc, ccb, hc_b, cB * 512);
    (void)hipMemsetAsync(s_s, 0, (long)cB * 4, stream);
    (void)hipMemsetAsync(rpool, 0, (long)cB * cH * 4, stream);
    k_s2s_score<<<(cN + 7) / 8, 256, 0, stream>>>(h, hc, batch, ae_s, s_s, cN);
    k_s2s_pool<<<(unsigned)(((long)cN * 512 + 255) / 256), 256, 0, stream>>>(
        h, ae_s, s_s, batch, rpool, cN);
    k_qstar<<<(cB * 1024 + 255) / 256, 256, 0, stream>>>(hc, rpool, q_star, qstar_b, cB * 1024);
  }

  // ---- output heads ----
  gemm(pooled_b, nullptr, cH, cH, nullptr, nullptr, 0, 0, nullptr, nullptr, 0, 0,
       lpW1_b, lp_b1, nullptr, nullptr, nullptr,
       nullptr, t1_b, nullptr, nullptr, cB, cH, F_SILU | F_STBF16);
  gemm(t1_b, nullptr, cH, cH, nullptr, nullptr, 0, 0, nullptr, nullptr, 0, 0,
       lpW2_b, lp_b2, nullptr, nullptr, nullptr,
       zloc, nullptr, nullptr, nullptr, cB, cLOC, F_STF32);
  gemm(qstar_b, nullptr, 1024, 1024, nullptr, nullptr, 0, 0, nullptr, nullptr, 0, 0,
       gpW1_b, gp_b1, nullptr, nullptr, nullptr,
       nullptr, t2_b, nullptr, nullptr, cB, cH, F_SILU | F_STBF16);
  gemm(t2_b, nullptr, cH, cH, nullptr, nullptr, 0, 0, nullptr, nullptr, 0, 0,
       gpW2_b, gp_b2, nullptr, nullptr, nullptr,
       zglob, nullptr, nullptr, nullptr, cB, cGLB, F_STF32);
}
